// MyRNN_11879879543319
// MI455X (gfx1250) — matrix operand order repaired and checked by reading, not compiled, not run on hardware
//
#include <hip/hip_runtime.h>

// Problem constants (from reference): B,T,V,E,U = 64,4995,4096,100,64
#define NB    64
#define NT    4995
#define NE    100
#define NU    64
#define G3    192    // 3*U gate width
#define KPAD  128    // layer-1 K (E=100) padded to multiple of 32
#define AS_S  136    // LDS row stride (halves) for phase-1 A staging (bank-spread, 16B aligned)
#define HH_S  72     // LDS row stride (halves) for f16 hidden state (16B aligned)
#define HP_S  196    // LDS row stride (floats) for gate buffers (bank-spread)

#if defined(__gfx1250__) && __has_builtin(__builtin_amdgcn_global_load_async_to_lds_b128)
#define USE_ASYNC_PF 1
#else
#define USE_ASYNC_PF 0
#endif

typedef _Float16 v16h __attribute__((ext_vector_type(16)));
typedef _Float16 v8h  __attribute__((ext_vector_type(8)));
typedef float    v8f  __attribute__((ext_vector_type(8)));
typedef int      v4i  __attribute__((vector_size(16)));   // matches async-builtin param

__device__ __forceinline__ v8f wmma16(v16h a, v16h b, v8f c) {
  // v_wmma_f32_16x16x32_f16: D = A(16x32) * B(32x16) + C
  return __builtin_amdgcn_wmma_f32_16x16x32_f16(false, a, false, b, (short)0, c,
                                                false, false);
}

// Build a B-operand tile (32x16) from a row-major f32 weight [K x 192].
// Layout: lane L holds column n = nt*16 + L%16, rows k = kc*32 + 8*(L/16) + (i&7) + 16*(i>>3).
__device__ __forceinline__ v16h load_b_f32(const float* __restrict__ w, int K,
                                           int nt, int kc, int lane) {
  const int half = lane >> 4;
  const int n = nt * 16 + (lane & 15);
  v16h b;
#pragma unroll
  for (int i = 0; i < 16; ++i) {
    const int k = kc * 32 + half * 8 + (i & 7) + 16 * (i >> 3);
    b[i] = (_Float16)((k < K) ? w[k * G3 + n] : 0.0f);
  }
  return b;
}

// Build an A-operand tile (16x32) from LDS row-major f16 (rowbase -> row m=lane%16).
// Lane layout: elems 0..7 -> K = kbase + 8*half + i ; elems 8..15 -> K = kbase + 16 + 8*half + i.
__device__ __forceinline__ v16h load_a_lds(const _Float16* rowbase, int kbase, int half) {
  const v8h lo = *(const v8h*)(rowbase + kbase + 8 * half);
  const v8h hi = *(const v8h*)(rowbase + kbase + 16 + 8 * half);
  v16h a;
#pragma unroll
  for (int i = 0; i < 8; ++i) { a[i] = lo[i]; a[i + 8] = hi[i]; }
  return a;
}

__device__ __forceinline__ float sigm(float x) { return 1.0f / (1.0f + __expf(-x)); }

__device__ __forceinline__ void wait_async_le2() {
#if USE_ASYNC_PF
#if __has_builtin(__builtin_amdgcn_s_wait_asynccnt)
  __builtin_amdgcn_s_wait_asynccnt(2);
#else
  asm volatile("s_wait_asynccnt 0x2" ::: "memory");
#endif
#endif
}

// ---------------------------------------------------------------------------
// Phase 1: xp1[b*T+t, 0:192] = emb[tokens[b,t]] @ W1 + b1[0]
// 12 waves/block, one 16-row x 192-col tile per iteration, grid-stride.
// ---------------------------------------------------------------------------
__global__ __launch_bounds__(384) void k_embed_proj(
    const int* __restrict__ tokens, const float* __restrict__ emb,
    const float* __restrict__ W1, const float* __restrict__ b1,
    float* __restrict__ xp1) {
  __shared__ __align__(16) _Float16 As[16 * AS_S];
  __shared__ int tok[16];

  const int tid  = threadIdx.x;
  const int lane = tid & 31;
  const int nt   = tid >> 5;       // wave id == N tile (0..11)
  const int half = lane >> 4;
  const int m    = lane & 15;

  // Hoist W1 B-tiles into registers for the whole kernel (reused every row-tile).
  v16h Bt[4];
#pragma unroll
  for (int kc = 0; kc < 4; ++kc) Bt[kc] = load_b_f32(W1, NE, nt, kc, lane);
  const float bias = b1[nt * 16 + m];  // b1[0][n]

  const int NROWT = (NB * NT) / 16;    // 19980 tiles of 16 gathered rows
  for (int rt = blockIdx.x; rt < NROWT; rt += gridDim.x) {
    __syncthreads();                   // protect As/tok reuse across iterations
    if (tid < 16) tok[tid] = tokens[rt * 16 + tid];
    __syncthreads();
    // Stage gathered A tile: 16 rows x 128 K (f32 -> f16, zero pad K>=100)
    for (int e = tid; e < 16 * KPAD; e += 384) {
      const int r = e >> 7, k = e & (KPAD - 1);
      const float v = (k < NE) ? emb[tok[r] * NE + k] : 0.0f;
      As[r * AS_S + k] = (_Float16)v;
    }
    __syncthreads();

    v8f acc;
#pragma unroll
    for (int r = 0; r < 8; ++r) acc[r] = bias;
#pragma unroll
    for (int kc = 0; kc < 4; ++kc) {
      const v16h a = load_a_lds(&As[m * AS_S], kc * 32, half);
      acc = wmma16(a, Bt[kc], acc);
    }
    // D: lane holds col n=m, rows r+8*half. xp1 is touch-once -> nontemporal.
#pragma unroll
    for (int r = 0; r < 8; ++r) {
      const size_t row = (size_t)rt * 16 + r + 8 * half;
      __builtin_nontemporal_store(acc[r], &xp1[row * G3 + nt * 16 + m]);
    }
  }
}

// ---------------------------------------------------------------------------
// Phase 2: fused 2-layer GRU scan. 4 blocks x 384 threads; block g owns batch
// rows g*16 .. g*16+15. All recurrent weights live in registers (6 v16h/wave).
// Next timestep's xp1 gate tile (12KB) is prefetched into LDS with async
// global->LDS copies (ASYNCcnt), double-buffered, overlapped with compute.
// ---------------------------------------------------------------------------
__global__ __launch_bounds__(384) void k_gru_scan(
    const float* __restrict__ xp1,
    const float* __restrict__ U1, const float* __restrict__ b1,
    const float* __restrict__ W2, const float* __restrict__ U2,
    const float* __restrict__ b2, const float* __restrict__ Wout,
    const float* __restrict__ bout, float* __restrict__ out) {
  __shared__ __align__(16) _Float16 h1h[16 * HH_S];
  __shared__ __align__(16) _Float16 h2h[16 * HH_S];
  __shared__ float h1f[16 * NU];
  __shared__ float h2f[16 * NU];
  __shared__ float hp1[16 * HP_S];
  __shared__ float hp2[16 * HP_S];
  __shared__ float xq2[16 * HP_S];
#if USE_ASYNC_PF
  __shared__ __align__(16) float pfb[2][16 * G3];
#endif

  const int tid  = threadIdx.x;
  const int lane = tid & 31;
  const int nt   = tid >> 5;       // wave id == N tile (0..11)
  const int half = lane >> 4;
  const int ln   = lane & 15;
  const int g    = blockIdx.x;     // batch group

  // Recurrent weight B-tiles in registers for the entire scan.
  v16h BU1[2], BW2[2], BU2[2];
#pragma unroll
  for (int kc = 0; kc < 2; ++kc) {
    BU1[kc] = load_b_f32(U1, NU, nt, kc, lane);
    BW2[kc] = load_b_f32(W2, NU, nt, kc, lane);
    BU2[kc] = load_b_f32(U2, NU, nt, kc, lane);
  }
  const float b1r = b1[G3 + nt * 16 + ln];  // b1[1][n]
  const float b2x = b2[nt * 16 + ln];       // b2[0][n]
  const float b2r = b2[G3 + nt * 16 + ln];  // b2[1][n]

  for (int e = tid; e < 16 * NU; e += 384) { h1f[e] = 0.0f; h2f[e] = 0.0f; }
  for (int e = tid; e < 16 * HH_S; e += 384) {
    h1h[e] = (_Float16)0.0f; h2h[e] = (_Float16)0.0f;
  }

#if USE_ASYNC_PF
  // Issue async prefetch of the xp1 tile for timestep tt into pfb[buf].
  // 16 rows x 192 floats = 768 16B chunks; 2 async-load instructions per wave.
  auto issue_pf = [&](int tt, int buf) {
#pragma unroll
    for (int q = 0; q < 2; ++q) {
      const int c = tid + q * 384;          // chunk id, 0..767
      const int m = c / 48;
      const int f = (c % 48) * 4;           // float offset within row
      const float* src = &xp1[((size_t)(g * 16 + m) * NT + tt) * G3 + f];
      float* dst = &pfb[buf][m * G3 + f];
      __builtin_amdgcn_global_load_async_to_lds_b128((v4i*)src, (v4i*)dst, 0, 0);
    }
  };
  issue_pf(0, 0);                           // prologue: prefetch t=0
#endif
  __syncthreads();

  for (int t = 0; t < NT; ++t) {
#if USE_ASYNC_PF
    // Kick off prefetch for t+1 (clamped; keeps wait-threshold logic uniform).
    const int tn = (t + 1 < NT) ? (t + 1) : (NT - 1);
    issue_pf(tn, (t + 1) & 1);
#endif
    // ---- layer-1 recurrence: hp1 = h1 @ U1 + b1[1] ----
    {
      v8f acc;
#pragma unroll
      for (int r = 0; r < 8; ++r) acc[r] = b1r;
#pragma unroll
      for (int kc = 0; kc < 2; ++kc) {
        const v16h a = load_a_lds(&h1h[ln * HH_S], kc * 32, half);
        acc = wmma16(a, BU1[kc], acc);
      }
#pragma unroll
      for (int r = 0; r < 8; ++r)
        hp1[(r + 8 * half) * HP_S + nt * 16 + ln] = acc[r];
    }
    // Oldest 2 async ops (this step's tile) must be done; the 2 just issued
    // for t+1 may remain in flight.
    wait_async_le2();
    __syncthreads();
    // ---- layer-1 gates ----
    for (int e = tid; e < 16 * NU; e += 384) {
      const int m = e >> 6, j = e & 63;
#if USE_ASYNC_PF
      const float* xrow = &pfb[t & 1][m * G3];
      const float xz = xrow[j];
      const float xr = xrow[64 + j];
      const float xh = xrow[128 + j];
#else
      const size_t ro = ((size_t)(g * 16 + m) * NT + t) * G3;
      const float xz = __builtin_nontemporal_load(&xp1[ro + j]);
      const float xr = __builtin_nontemporal_load(&xp1[ro + 64 + j]);
      const float xh = __builtin_nontemporal_load(&xp1[ro + 128 + j]);
#endif
      const float pz = hp1[m * HP_S + j];
      const float pr = hp1[m * HP_S + 64 + j];
      const float ph = hp1[m * HP_S + 128 + j];
      const float z  = sigm(xz + pz);
      const float r  = sigm(xr + pr);
      const float hh = tanhf(xh + r * ph);
      const float hn = z * h1f[e] + (1.0f - z) * hh;
      h1f[e] = hn;
      h1h[m * HH_S + j] = (_Float16)hn;
    }
    __syncthreads();
    // ---- layer-2: hp2 = h2 @ U2 + b2[1] ; xq2 = h1_new @ W2 + b2[0] ----
    {
      v8f a2, a3;
#pragma unroll
      for (int r = 0; r < 8; ++r) { a2[r] = b2r; a3[r] = b2x; }
#pragma unroll
      for (int kc = 0; kc < 2; ++kc) {
        const v16h ah2 = load_a_lds(&h2h[ln * HH_S], kc * 32, half);
        a2 = wmma16(ah2, BU2[kc], a2);
        const v16h ah1 = load_a_lds(&h1h[ln * HH_S], kc * 32, half);
        a3 = wmma16(ah1, BW2[kc], a3);
      }
#pragma unroll
      for (int r = 0; r < 8; ++r) {
        const int ro = (r + 8 * half) * HP_S + nt * 16 + ln;
        hp2[ro] = a2[r];
        xq2[ro] = a3[r];
      }
    }
    __syncthreads();
    // ---- layer-2 gates ----
    for (int e = tid; e < 16 * NU; e += 384) {
      const int m = e >> 6, j = e & 63;
      const float z  = sigm(xq2[m * HP_S + j] + hp2[m * HP_S + j]);
      const float r  = sigm(xq2[m * HP_S + 64 + j] + hp2[m * HP_S + 64 + j]);
      const float hh = tanhf(xq2[m * HP_S + 128 + j] + r * hp2[m * HP_S + 128 + j]);
      const float hn = z * h2f[e] + (1.0f - z) * hh;
      h2f[e] = hn;
      h2h[m * HH_S + j] = (_Float16)hn;
    }
    __syncthreads();
  }

  // ---- head: sigmoid(h2 @ Wout + bout) ----
  if (tid < 16) {
    float s = bout[0];
#pragma unroll 8
    for (int j = 0; j < NU; ++j) s += h2f[tid * NU + j] * Wout[j];
    out[g * 16 + tid] = sigm(s);
  }
}

extern "C" void kernel_launch(void* const* d_in, const int* in_sizes, int n_in,
                              void* d_out, int out_size, void* d_ws, size_t ws_size,
                              hipStream_t stream) {
  const int*   tokens = (const int*)d_in[0];
  const float* emb    = (const float*)d_in[1];
  const float* W1     = (const float*)d_in[2];
  const float* U1     = (const float*)d_in[3];
  const float* b1     = (const float*)d_in[4];
  const float* W2     = (const float*)d_in[5];
  const float* U2     = (const float*)d_in[6];
  const float* b2     = (const float*)d_in[7];
  const float* Wout   = (const float*)d_in[8];
  const float* bout   = (const float*)d_in[9];
  float* out = (float*)d_out;

  float* xp1 = (float*)d_ws;  // [B*T, 192] f32 (~245 MB)

  k_embed_proj<<<1024, 384, 0, stream>>>(tokens, emb, W1, b1, xp1);
  k_gru_scan<<<4, 384, 0, stream>>>(xp1, U1, b1, W2, U2, b2, Wout, bout, out);
}
